// GMMNet_24068996727341
// MI455X (gfx1250) — matrix operand order repaired and checked by reading, not compiled, not run on hardware
//
#include <hip/hip_runtime.h>
#include <hip/hip_bf16.h>

// CDNA5 / gfx1250: wave32, WMMA 16x16x32 f16->f32.
typedef __attribute__((ext_vector_type(16))) _Float16 v16h;
typedef __attribute__((ext_vector_type(8)))  float    v8f;

#define H64   64
#define KH    256   // K*H
#define WCOLS 320   // K*H + H  (g | root fused)

// ---------------- embed: x[N,16] -> h0[Npad,64] (one-hot(31) + cont(8) + pad) --
__global__ void embed_kernel(const float* __restrict__ x, float* __restrict__ h0,
                             int N, int Npad) {
  int n = blockIdx.x * blockDim.x + threadIdx.x;
  if (n >= Npad) return;
  float* row = h0 + (size_t)n * H64;
  #pragma unroll
  for (int j = 0; j < H64; ++j) row[j] = 0.f;
  if (n >= N) return;
  const float* xr = x + (size_t)n * 16;
  const int offs[8] = {0, 7, 12, 16, 20, 22, 24, 28};  // cumsum of FEATS[:-1]
  #pragma unroll
  for (int i = 0; i < 8; ++i) {
    int v = (int)xr[i];
    row[offs[i] + v] = 1.f;
  }
  #pragma unroll
  for (int i = 0; i < 8; ++i) row[31 + i] = xr[8 + i];
}

// ---------------- pack weights into WMMA B-register swizzled layout ------------
// Bsw element layout: out[((nt*ksteps + ks)*32 + lane)*16 + j] holds
//   B[k][c] with k = ks*32 + (lane>>4)*8 + (j&7) + (j>>3)*16,  c = nt*16 + (lane&15)
// so a lane's 16 halfs for one WMMA are contiguous (two b128 loads).
// Source: fused [in_rows x 320] = g|root when root != nullptr, else g[in_rows x ncols].
__global__ void pack_bsw_kernel(const float* __restrict__ g,
                                const float* __restrict__ root,
                                int in_rows, int ncols, int ksteps,
                                _Float16* __restrict__ out, int total) {
  int idx = blockIdx.x * blockDim.x + threadIdx.x;
  if (idx >= total) return;
  int j    = idx & 15;
  int lane = (idx >> 4) & 31;
  int rest = idx >> 9;
  int ks   = rest % ksteps;
  int nt   = rest / ksteps;
  int k = ks * 32 + (lane >> 4) * 8 + (j & 7) + (j >> 3) * 16;
  int c = nt * 16 + (lane & 15);
  float v = 0.f;
  if (k < in_rows) {
    if (root) v = (c < KH) ? g[(size_t)k * KH + c] : root[(size_t)k * H64 + (c - KH)];
    else      v = g[(size_t)k * ncols + c];
  }
  out[idx] = (_Float16)v;
}

// ---------------- A-tile loader: 16x32 f32 -> f16 WMMA A layout ----------------
// p = row base + (lane>>4)*8; loads K = {kb..kb+7, 16+kb..16+kb+7}
__device__ __forceinline__ v16h load_a_tile(const float* __restrict__ p) {
  float4 f0 = *(const float4*)(p);
  float4 f1 = *(const float4*)(p + 4);
  float4 f2 = *(const float4*)(p + 16);
  float4 f3 = *(const float4*)(p + 20);
  v16h a;
  a[0]  = (_Float16)f0.x; a[1]  = (_Float16)f0.y; a[2]  = (_Float16)f0.z; a[3]  = (_Float16)f0.w;
  a[4]  = (_Float16)f1.x; a[5]  = (_Float16)f1.y; a[6]  = (_Float16)f1.z; a[7]  = (_Float16)f1.w;
  a[8]  = (_Float16)f2.x; a[9]  = (_Float16)f2.y; a[10] = (_Float16)f2.z; a[11] = (_Float16)f2.w;
  a[12] = (_Float16)f3.x; a[13] = (_Float16)f3.y; a[14] = (_Float16)f3.z; a[15] = (_Float16)f3.w;
  return a;
}

// ---------------- WMMA GEMM: C[f32] = A[f32->f16] * Bsw[f16, swizzled] ---------
// grid.x covers M/(16*MREP) tile groups; (blockIdx.y*4 + wave) = N/16 tile.
// blockDim = 128 (4 waves). KSIZE compile-time => fully unrolled K loop.
// Row/B base pointers hoisted out of the K loop so k-steps become immediate
// offsets on the b128 loads (ks*128B for A, ks*1024B for B).
template <int KSIZE, int MREP>
__global__ void gemm_wmma_kernel(const float* __restrict__ A,
                                 const _Float16* __restrict__ Bsw,
                                 float* __restrict__ C,
                                 int lda, int ldc, int ntiles) {
  constexpr int KSTEPS = KSIZE / 32;
  int lane  = threadIdx.x & 31;
  int wid   = threadIdx.x >> 5;
  int ntile = blockIdx.y * 4 + wid;
  if (ntile >= ntiles) return;  // wave-uniform exit; EXEC stays all-ones
  int mbase = blockIdx.x * (MREP * 16);

  const float* rowp[MREP];
  #pragma unroll
  for (int r = 0; r < MREP; ++r)
    rowp[r] = A + (size_t)(mbase + r * 16 + (lane & 15)) * lda + (lane >> 4) * 8;
  const _Float16* bp = Bsw + (((size_t)ntile * KSTEPS) * 32 + lane) * 16;

  v8f acc[MREP];
  #pragma unroll
  for (int r = 0; r < MREP; ++r) acc[r] = (v8f){};

  #pragma unroll
  for (int ks = 0; ks < KSTEPS; ++ks) {
    v16h b = *(const v16h*)(bp + ks * (32 * 16));
    #pragma unroll
    for (int r = 0; r < MREP; ++r) {
      v16h a = load_a_tile(rowp[r] + ks * 32);
      acc[r] = __builtin_amdgcn_wmma_f32_16x16x32_f16(false, a, false, b,
                                                      (short)0, acc[r], false, false);
    }
  }
  int col = ntile * 16 + (lane & 15);
  #pragma unroll
  for (int r = 0; r < MREP; ++r) {
    int rbase = mbase + r * 16 + 8 * (lane >> 4);
    float* Cp = C + (size_t)rbase * ldc + col;
    #pragma unroll
    for (int v = 0; v < 8; ++v) Cp[(size_t)v * ldc] = acc[r][v];
  }
}

// ---------------- per-edge Gaussian mixture weights w[E,4] ---------------------
__global__ void edge_w_kernel(const float* __restrict__ ea,
                              const float* __restrict__ mu,
                              const float* __restrict__ sigma,
                              float* __restrict__ w, int E) {
  int e = blockIdx.x * blockDim.x + threadIdx.x;
  if (e >= E) return;
  float u0 = ea[e * 3 + 0], u1 = ea[e * 3 + 1], u2 = ea[e * 3 + 2];
  #pragma unroll
  for (int k = 0; k < 4; ++k) {
    float s = 0.f;
    float d0 = u0 - mu[k * 3 + 0], g0 = sigma[k * 3 + 0];
    float d1 = u1 - mu[k * 3 + 1], g1 = sigma[k * 3 + 1];
    float d2 = u2 - mu[k * 3 + 2], g2 = sigma[k * 3 + 2];
    s += d0 * d0 / (1e-15f + g0 * g0);
    s += d1 * d1 / (1e-15f + g1 * g1);
    s += d2 * d2 / (1e-15f + g2 * g2);
    w[e * 4 + k] = __expf(-0.5f * s);
  }
}

// ---------------- in-degree counts (once; edge list is layer-invariant) --------
__global__ void edge_cnt_kernel(const int* __restrict__ dst,
                                float* __restrict__ cnt, int E) {
  int e = blockIdx.x * blockDim.x + threadIdx.x;
  if (e >= E) return;
  atomicAdd(&cnt[dst[e]], 1.f);
}

// ---------------- edge scatter: agg[dst,h] += sum_k w[e,k]*proj[src,k*64+h] ----
__global__ void edge_agg_kernel(const int* __restrict__ src,
                                const int* __restrict__ dst,
                                const float* __restrict__ w,
                                const float* __restrict__ proj,
                                float* __restrict__ agg, int E) {
  long long i = (long long)blockIdx.x * blockDim.x + threadIdx.x;
  if (i >= (long long)E * 64) return;
  int e = (int)(i >> 6), h = (int)(i & 63);
  int s = src[e], d = dst[e];
  const float* pr = proj + (size_t)s * WCOLS + h;
  float4 we = *(const float4*)(w + (size_t)e * 4);
  float val = we.x * pr[0] + we.y * pr[64] + we.z * pr[128] + we.w * pr[192];
  atomicAdd(&agg[(size_t)d * 64 + h], val);
}

// ---------------- y = agg/max(cnt,1) + root-projection (bias cancels in BN) ----
__global__ void finalize_y_kernel(float* __restrict__ agg,
                                  const float* __restrict__ proj,
                                  const float* __restrict__ cnt, int N) {
  long long i = (long long)blockIdx.x * blockDim.x + threadIdx.x;
  if (i >= (long long)N * 64) return;
  int n = (int)(i >> 6), h = (int)(i & 63);
  float c = cnt[n]; c = c < 1.f ? 1.f : c;
  agg[i] = agg[i] / c + proj[(size_t)n * WCOLS + KH + h];
}

// ---------------- per-column sums / sums-of-squares (64 cols) ------------------
__global__ void colstats_kernel(const float* __restrict__ Y, long long total,
                                float* __restrict__ sums, float* __restrict__ sumsq) {
  __shared__ float ss[64], sq[64];
  if (threadIdx.x < 64) { ss[threadIdx.x] = 0.f; sq[threadIdx.x] = 0.f; }
  __syncthreads();
  for (long long i = (long long)blockIdx.x * blockDim.x + threadIdx.x; i < total;
       i += (long long)gridDim.x * blockDim.x) {
    float v = Y[i];
    int c = (int)(i & 63);
    atomicAdd(&ss[c], v);
    atomicAdd(&sq[c], v * v);
  }
  __syncthreads();
  if (threadIdx.x < 64) {
    atomicAdd(&sums[threadIdx.x], ss[threadIdx.x]);
    atomicAdd(&sumsq[threadIdx.x], sq[threadIdx.x]);
  }
}

// ---------------- batch-norm + (leaky-)relu; zero-pads rows >= rows ------------
__global__ void bn_apply_kernel(const float* __restrict__ Y,
                                const float* __restrict__ sums,
                                const float* __restrict__ sumsq,
                                const float* __restrict__ gamma,
                                const float* __restrict__ beta,
                                float* __restrict__ out,
                                int rows, int rowsOut, float alpha) {
  int i = blockIdx.x * blockDim.x + threadIdx.x;
  if (i >= rowsOut * 64) return;
  int n = i >> 6, c = i & 63;
  float o = 0.f;
  if (n < rows) {
    float invn = 1.f / (float)rows;
    float m = sums[c] * invn;
    float v = sumsq[c] * invn - m * m;
    float val = (Y[i] - m) * rsqrtf(v + 1e-5f) * gamma[c] + beta[c];
    o = val > 0.f ? val : alpha * val;
  }
  out[i] = o;
}

__global__ void add_kernel(const float* __restrict__ a, const float* __restrict__ b,
                           float* __restrict__ o, int n) {
  int i = blockIdx.x * blockDim.x + threadIdx.x;
  if (i < n) o[i] = a[i] + b[i];
}

// ---------------- segment-max pooling ------------------------------------------
__global__ void pool_init_kernel(float* __restrict__ pooled, int n) {
  int i = blockIdx.x * blockDim.x + threadIdx.x;
  if (i < n) pooled[i] = __int_as_float(0xFF800000);  // -inf
}

__device__ __forceinline__ void atomicMaxF(float* addr, float val) {
  if (val >= 0.f) atomicMax((int*)addr, __float_as_int(val));
  else            atomicMin((unsigned int*)addr, __float_as_uint(val));
}

__global__ void pool_kernel(const float* __restrict__ xt, const float* __restrict__ x1,
                            const float* __restrict__ x2, const float* __restrict__ x3,
                            const int* __restrict__ bidx, float* __restrict__ pooled,
                            int N) {
  long long i = (long long)blockIdx.x * blockDim.x + threadIdx.x;
  if (i >= (long long)N * 256) return;
  int n = (int)(i >> 8), c = (int)(i & 255);
  float v;
  if      (c < 64)  v = xt[(size_t)n * 64 + c];
  else if (c < 128) v = x1[(size_t)n * 64 + (c - 64)];
  else if (c < 192) v = x2[(size_t)n * 64 + (c - 128)];
  else              v = x3[(size_t)n * 64 + (c - 192)];
  int g = bidx[n];
  atomicMaxF(&pooled[(size_t)g * 256 + c], v);
}

// ---------------- final head: out[g] = z2[g,:] . W2 + b2 -----------------------
__global__ void head_kernel(const float* __restrict__ z2, const float* __restrict__ W2,
                            const float* __restrict__ b2, float* __restrict__ out, int G) {
  int g = blockIdx.x * blockDim.x + threadIdx.x;
  if (g >= G) return;
  float acc = b2[0];
  #pragma unroll 8
  for (int h = 0; h < 64; ++h) acc += z2[(size_t)g * 64 + h] * W2[h];
  out[g] = acc;
}

// ================================================================================
extern "C" void kernel_launch(void* const* d_in, const int* in_sizes, int n_in,
                              void* d_out, int out_size, void* d_ws, size_t ws_size,
                              hipStream_t stream) {
  const float* x     = (const float*)d_in[0];
  const int*   eidx  = (const int*)  d_in[1];   // [2,E]: row0=src, row1=dst
  const int*   bidx  = (const int*)  d_in[2];
  const float* eattr = (const float*)d_in[3];

  const int N = in_sizes[0] / 16;
  const int E = in_sizes[1] / 2;
  const int G = out_size;
  const int Npad = (N + 63) & ~63;   // multiple of 64 rows => MREP=4 tile groups
  const int Gpad = (G + 63) & ~63;
  const int* srcp = eidx;
  const int* dstp = eidx + E;

  // Param leaves in setup_inputs() insertion order:
  // 4..10 init{g,mu,sigma,root,bias,bn_g,bn_b}, 11..17 head, 18..24 body,
  // 25..31 tail, 32 W1, 33 b1, 34 bn_g, 35 bn_b, 36 W2, 37 b2, 38 num_graphs
  const float* P[40];
  for (int i = 0; i < n_in && i < 40; ++i) P[i] = (const float*)d_in[i];

  // ---- carve workspace ----
  char* wsp = (char*)d_ws;
  size_t off = 0;
  auto carve = [&](size_t bytes) -> void* {
    void* p = wsp + off;
    off += (bytes + 255) & ~(size_t)255;
    return p;
  };
  float*    h0     = (float*)carve((size_t)Npad * 64 * 4);
  float*    h1     = (float*)carve((size_t)Npad * 64 * 4);
  float*    x1     = (float*)carve((size_t)Npad * 64 * 4);
  float*    x2     = (float*)carve((size_t)Npad * 64 * 4);
  float*    x3     = (float*)carve((size_t)Npad * 64 * 4);
  float*    xt     = (float*)carve((size_t)Npad * 64 * 4);
  float*    proj   = (float*)carve((size_t)Npad * WCOLS * 4);
  float*    w4     = (float*)carve((size_t)E * 4 * 4);
  float*    cnt    = (float*)carve((size_t)N * 4);
  float*    agg    = (float*)carve((size_t)N * 64 * 4);
  float*    sums   = (float*)carve(64 * 4);
  float*    sumsq  = (float*)carve(64 * 4);
  _Float16* Bsw    = (_Float16*)carve((size_t)20 * 2 * 32 * 16 * 2);  // 20 ntiles, K=64
  _Float16* W1sw   = (_Float16*)carve((size_t)4 * 8 * 32 * 16 * 2);   // 4 ntiles,  K=256
  float*    pooled = (float*)carve((size_t)Gpad * 256 * 4);
  float*    z      = (float*)carve((size_t)Gpad * 64 * 4);
  float*    z2     = (float*)carve((size_t)G * 64 * 4);
  (void)ws_size;

  const long long eh  = (long long)E * 64;
  const long long nh  = (long long)N * 64;
  const int       nph = Npad * 64;
  const int bsw_total  = 20 * 2 * 32 * 16;  // ntiles*ksteps*32*16
  const int w1sw_total = 4 * 8 * 32 * 16;

  // ---- embed + degree counts (once) ----
  embed_kernel<<<(Npad + 127) / 128, 128, 0, stream>>>(x, h0, N, Npad);
  hipMemsetAsync(cnt, 0, (size_t)N * 4, stream);
  edge_cnt_kernel<<<(E + 255) / 256, 256, 0, stream>>>(dstp, cnt, E);

  // ---- one GMM conv layer ----
  auto conv_layer = [&](const float* hin, const float* const* L, int in_ch, float* hout) {
    // L: {g, mu, sigma, root, bias, bn_g, bn_b}
    pack_bsw_kernel<<<(bsw_total + 255) / 256, 256, 0, stream>>>(L[0], L[3], in_ch, WCOLS,
                                                                 2, Bsw, bsw_total);
    edge_w_kernel<<<(E + 255) / 256, 256, 0, stream>>>(eattr, L[1], L[2], w4, E);
    dim3 gg(Npad / 64, 5);  // 4 M-tiles/wave, 20 N-tiles over 5*4 waves
    gemm_wmma_kernel<64, 4><<<gg, 128, 0, stream>>>(hin, Bsw, proj, 64, WCOLS, 20);
    hipMemsetAsync(agg, 0, (size_t)N * 64 * 4, stream);
    edge_agg_kernel<<<(unsigned)((eh + 255) / 256), 256, 0, stream>>>(srcp, dstp, w4, proj, agg, E);
    finalize_y_kernel<<<(unsigned)((nh + 255) / 256), 256, 0, stream>>>(agg, proj, cnt, N);
    hipMemsetAsync(sums, 0, 64 * 4, stream);
    hipMemsetAsync(sumsq, 0, 64 * 4, stream);
    colstats_kernel<<<512, 256, 0, stream>>>(agg, nh, sums, sumsq);
    bn_apply_kernel<<<(nph + 255) / 256, 256, 0, stream>>>(agg, sums, sumsq, L[5], L[6],
                                                           hout, N, Npad, 0.01f);
  };

  const float* Linit[7], *Lhead[7], *Lbody[7], *Ltail[7];
  for (int j = 0; j < 7; ++j) {
    Linit[j] = P[4 + j];
    Lhead[j] = P[11 + j];
    Lbody[j] = P[18 + j];
    Ltail[j] = P[25 + j];
  }

  conv_layer(h0, Linit, 39, h1);   // h  = init(embed)
  conv_layer(h1, Lhead, 64, x1);   // x1 = head(h)
  conv_layer(x1, Lbody, 64, x2);   // x2 = body(x1)
  add_kernel<<<(nph + 255) / 256, 256, 0, stream>>>(h1, x2, x3, nph);  // x3 = h + x2
  conv_layer(x3, Ltail, 64, xt);   // xt = tail(x3)

  // ---- segment-max pool over cat[xt,x1,x2,x3] ----
  pool_init_kernel<<<(Gpad * 256 + 255) / 256, 256, 0, stream>>>(pooled, Gpad * 256);
  pool_kernel<<<(unsigned)(((long long)N * 256 + 255) / 256), 256, 0, stream>>>(
      xt, x1, x2, x3, bidx, pooled, N);

  // ---- MLP head: z = pooled @ W1 (b1 cancels in BN) ----
  pack_bsw_kernel<<<(w1sw_total + 255) / 256, 256, 0, stream>>>(P[32], nullptr, 256, 64,
                                                                8, W1sw, w1sw_total);
  dim3 g2(Gpad / 64, 1);  // 4 M-tiles/wave, 4 N-tiles over 4 waves
  gemm_wmma_kernel<256, 4><<<g2, 128, 0, stream>>>(pooled, W1sw, z, 256, 64, 4);
  hipMemsetAsync(sums, 0, 64 * 4, stream);
  hipMemsetAsync(sumsq, 0, 64 * 4, stream);
  colstats_kernel<<<128, 256, 0, stream>>>(z, (long long)G * 64, sums, sumsq);
  bn_apply_kernel<<<(G * 64 + 255) / 256, 256, 0, stream>>>(z, sums, sumsq, P[34], P[35],
                                                            z2, G, G, 0.0f);  // relu
  head_kernel<<<(G + 127) / 128, 128, 0, stream>>>(z2, P[36], P[37], (float*)d_out, G);
}
